// Linear_73478300500409
// MI455X (gfx1250) — matrix operand order repaired and checked by reading, not compiled, not run on hardware
//
#include <hip/hip_runtime.h>
#include <hip/hip_bf16.h>

typedef __attribute__((ext_vector_type(16))) __bf16 v16bf;
typedef __attribute__((ext_vector_type(8)))  __bf16 v8bf;
typedef __attribute__((ext_vector_type(8)))  float  v8f;
typedef __attribute__((ext_vector_type(4)))  unsigned int v4u;
typedef __attribute__((ext_vector_type(8)))  int v8i;
typedef __attribute__((ext_vector_type(4)))  int v4i;

#define D_IN   4096
#define D_OUT  4096
#define M_TOT  8192
#define RANK   64
#define SCALE_F 0.25f

#define BM 128
#define BN 128
#define BK 64
#define LSTR 72   // ushort stride per LDS row = 64 data + 8 pad (144B)

__device__ __forceinline__ unsigned int f2bf_u(float f) {
  union { float f; unsigned int u; } v; v.f = f;
  unsigned int r = v.u + 0x7FFFu + ((v.u >> 16) & 1u);  // round-nearest-even
  return r >> 16;
}
__device__ __forceinline__ unsigned int pk2(float a, float b) {
  return f2bf_u(a) | (f2bf_u(b) << 16);
}

// ---- Tensor Data Mover: DMA a [128 rows x 64 bf16] tile into LDS with
// hardware row padding (32 DWORDs data + 4 DWORDs pad == LSTR of 72 shorts).
__device__ __forceinline__ void tdm_load_tile(unsigned lds_off,
                                              const unsigned short* gptr,
                                              unsigned stride_elems) {
  unsigned long long ga = (unsigned long long)(size_t)gptr;
  v4u g0;
  g0[0] = 1u;                                   // count=1, no gather
  g0[1] = lds_off;                              // lds_addr (bytes)
  g0[2] = (unsigned)ga;                         // global_addr[31:0]
  g0[3] = (unsigned)((ga >> 32) & 0x01FFFFFFu)  // global_addr[56:32]
          | (2u << 30);                         // type=2 (image)
  v8i g1;
  g1[0] = (int)((1u << 16)      // data_size = 2 bytes
              | (1u << 20)      // pad_enable
              | (4u << 22)      // pad_interval: 32 DWORDs
              | (3u << 25));    // pad_amount : 4 DWORDs
  g1[1] = (int)((64u & 0xFFFFu) << 16);                  // tensor_dim0 = 64 (lo16)
  g1[2] = (int)(((64u >> 16) & 0xFFFFu) | (128u << 16)); // tensor_dim0 hi, tensor_dim1 = 128
  g1[3] = (int)(0u | (64u << 16));                       // tensor_dim1 hi, tile_dim0 = 64
  g1[4] = (int)(128u);                                   // tile_dim1 = 128, tile_dim2 = 0
  g1[5] = (int)stride_elems;                             // tensor_dim0_stride lo
  g1[6] = 0;                                             // stride hi, dim1_stride lo
  g1[7] = 0;
  v4i z4 = {0, 0, 0, 0};
  v8i z8 = {0, 0, 0, 0, 0, 0, 0, 0};
  __builtin_amdgcn_tensor_load_to_lds(g0, g1, z4, z4, z8, 0);
}

// ---------------- prep kernels ----------------
__global__ void __launch_bounds__(256) cvt_bf16_kernel(
    const float* __restrict__ src, unsigned int* __restrict__ dst, int n4) {
  int i = blockIdx.x * blockDim.x + threadIdx.x;
  if (i >= n4) return;
  float4 v = ((const float4*)src)[i];
  ((uint2*)dst)[i] = make_uint2(pk2(v.x, v.y), pk2(v.z, v.w));
}

__global__ void __launch_bounds__(256) addcvt_bf16_kernel(
    const float* __restrict__ p, const float* __restrict__ q,
    unsigned int* __restrict__ dst, int n4) {
  int i = blockIdx.x * blockDim.x + threadIdx.x;
  if (i >= n4) return;
  float4 a = ((const float4*)p)[i];
  float4 b = ((const float4*)q)[i];
  ((uint2*)dst)[i] = make_uint2(pk2(a.x + b.x, a.y + b.y), pk2(a.z + b.z, a.w + b.w));
}

// ---------------- rank-64 LoRA "A" projection: av = SCALE * x @ Aw^T ----------------
__global__ void __launch_bounds__(256) lora_a_kernel(
    const float* __restrict__ x, const unsigned short* __restrict__ Aw,
    float* __restrict__ av) {
  const int tid  = threadIdx.x;
  const int wave = tid >> 5, lane = tid & 31;
  const int half = lane >> 4, l16 = lane & 15;
  const long row = (long)blockIdx.x * 128 + wave * 16 + l16;

  v8f acc[4] = {};
  const float* xr = x + row * D_IN + half * 8;
  for (int k = 0; k < D_IN; k += 32) {
    float4 c0 = *(const float4*)(xr + k);
    float4 c1 = *(const float4*)(xr + k + 4);
    float4 c2 = *(const float4*)(xr + k + 16);
    float4 c3 = *(const float4*)(xr + k + 20);
    union { v16bf v; unsigned int u[8]; } a;
    a.u[0] = pk2(c0.x, c0.y); a.u[1] = pk2(c0.z, c0.w);
    a.u[2] = pk2(c1.x, c1.y); a.u[3] = pk2(c1.z, c1.w);
    a.u[4] = pk2(c2.x, c2.y); a.u[5] = pk2(c2.z, c2.w);
    a.u[6] = pk2(c3.x, c3.y); a.u[7] = pk2(c3.z, c3.w);
#pragma unroll
    for (int nt = 0; nt < 4; ++nt) {
      const unsigned short* bp = Aw + (long)(nt * 16 + l16) * D_IN + k + half * 8;
      v8bf lo = *(const v8bf*)bp;
      v8bf hi = *(const v8bf*)(bp + 16);
      v16bf b = __builtin_shufflevector(lo, hi, 0,1,2,3,4,5,6,7,8,9,10,11,12,13,14,15);
      acc[nt] = __builtin_amdgcn_wmma_f32_16x16x32_bf16(
          false, a.v, false, b, (short)0, acc[nt], false, false);
    }
  }
  const long rb = (long)blockIdx.x * 128 + wave * 16 + (half ? 8 : 0);
#pragma unroll
  for (int nt = 0; nt < 4; ++nt)
#pragma unroll
    for (int i = 0; i < 8; ++i)
      av[(rb + i) * RANK + nt * 16 + l16] = acc[nt][i] * SCALE_F;
}

// ---------------- main fused GEMM: out = x@W^T + bias + av@Bw^T ----------------
__global__ void __launch_bounds__(256) main_gemm_kernel(
    const float* __restrict__ x, const unsigned short* __restrict__ wb,
    const float* __restrict__ bias, const float* __restrict__ av,
    const unsigned short* __restrict__ bw, float* __restrict__ out) {
  __shared__ unsigned short sx[BM * LSTR];
  __shared__ unsigned short sw[2][BN * LSTR];   // TDM double buffer

  const int tid = threadIdx.x;
  const int n0  = blockIdx.x * BN;
  const long m0 = (long)blockIdx.y * BM;

  const int wave = tid >> 5, lane = tid & 31;
  const int wm0  = (wave >> 2) * 64;   // 0 / 64
  const int wn0  = (wave & 3) * 32;    // 0,32,64,96
  const int half = lane >> 4, l16 = lane & 15;

  const int xrow = tid >> 4;           // 0..15
  const int xcol = (tid & 15) * 4;     // 0..60

  float4 xr[8];
  const int NIT = D_IN / BK + 1;       // 64 main slabs + 1 fused LoRA slab

  auto loadX = [&](int it) {
    const float* xsrc; long xstr;
    if (it < D_IN / BK) { xsrc = x + it * BK; xstr = D_IN; }
    else                { xsrc = av;          xstr = RANK; }   // rank == BK
#pragma unroll
    for (int p = 0; p < 8; ++p)
      xr[p] = *(const float4*)(xsrc + (m0 + xrow + p * 16) * xstr + xcol);
  };
  auto wTile = [&](int it) -> const unsigned short* {
    return (it < D_IN / BK) ? (wb + (long)n0 * D_IN + it * BK)
                            : (bw + (long)n0 * RANK);
  };
  auto wStride = [&](int it) -> unsigned {
    return (it < D_IN / BK) ? (unsigned)D_IN : (unsigned)RANK;
  };

  v8f acc[4][2] = {};

  if (wave == 0)
    tdm_load_tile((unsigned)(size_t)(void*)&sw[0][0], wTile(0), wStride(0));
  loadX(0);

  for (int it = 0; it < NIT; ++it) {
    __syncthreads();                        // all waves done reading sx / sw[(it-1)&1]
#pragma unroll
    for (int p = 0; p < 8; ++p) {
      uint2 o = make_uint2(pk2(xr[p].x, xr[p].y), pk2(xr[p].z, xr[p].w));
      *(uint2*)&sx[(xrow + p * 16) * LSTR + xcol] = o;
    }
    if (wave == 0) {
      if (it + 1 < NIT) {
        tdm_load_tile((unsigned)(size_t)(void*)&sw[(it + 1) & 1][0],
                      wTile(it + 1), wStride(it + 1));
        __builtin_amdgcn_s_wait_tensorcnt(1);   // slab `it` DMA complete
      } else {
        __builtin_amdgcn_s_wait_tensorcnt(0);
      }
    }
    __syncthreads();                        // sx staged + sw[it&1] DMA visible

    if (it + 1 < NIT) loadX(it + 1);        // overlap next x fetch with WMMAs

    const unsigned short* swb = &sw[it & 1][0];
#pragma unroll
    for (int kc = 0; kc < BK; kc += 32) {
      v16bf afr[4], bfr[2];
#pragma unroll
      for (int mt = 0; mt < 4; ++mt) {
        const unsigned short* s = &sx[(wm0 + mt * 16 + l16) * LSTR + kc + half * 8];
        v8bf lo = *(const v8bf*)s;
        v8bf hi = *(const v8bf*)(s + 16);
        afr[mt] = __builtin_shufflevector(lo, hi, 0,1,2,3,4,5,6,7,8,9,10,11,12,13,14,15);
      }
#pragma unroll
      for (int nt = 0; nt < 2; ++nt) {
        const unsigned short* s = &swb[(wn0 + nt * 16 + l16) * LSTR + kc + half * 8];
        v8bf lo = *(const v8bf*)s;
        v8bf hi = *(const v8bf*)(s + 16);
        bfr[nt] = __builtin_shufflevector(lo, hi, 0,1,2,3,4,5,6,7,8,9,10,11,12,13,14,15);
      }
#pragma unroll
      for (int mt = 0; mt < 4; ++mt)
#pragma unroll
        for (int nt = 0; nt < 2; ++nt)
          acc[mt][nt] = __builtin_amdgcn_wmma_f32_16x16x32_bf16(
              false, afr[mt], false, bfr[nt], (short)0, acc[mt][nt], false, false);
    }
  }

  // epilogue: bias add + store
#pragma unroll
  for (int nt = 0; nt < 2; ++nt) {
    const int col = n0 + wn0 + nt * 16 + l16;
    const float bb = bias[col];
#pragma unroll
    for (int mt = 0; mt < 4; ++mt) {
      const long r0 = m0 + wm0 + mt * 16 + (half ? 8 : 0);
#pragma unroll
      for (int i = 0; i < 8; ++i)
        out[(r0 + i) * (long)D_OUT + col] = acc[mt][nt][i] + bb;
    }
  }
}

// ---------------- launch ----------------
extern "C" void kernel_launch(void* const* d_in, const int* in_sizes, int n_in,
                              void* d_out, int out_size, void* d_ws, size_t ws_size,
                              hipStream_t stream) {
  const float* x      = (const float*)d_in[0];
  const float* W_base = (const float*)d_in[1];
  const float* b_base = (const float*)d_in[2];
  const float* A      = (const float*)d_in[3];
  const float* B      = (const float*)d_in[4];
  const float* WL     = (const float*)d_in[5];
  const float* WR     = (const float*)d_in[6];
  float* out = (float*)d_out;

  char* ws = (char*)d_ws;
  unsigned short* wbf  = (unsigned short*)(ws);                 // 33,554,432 B  W_base bf16
  unsigned short* awbf = (unsigned short*)(ws + 33554432);      //    524,288 B  (A+WL) bf16
  unsigned short* bwbf = (unsigned short*)(ws + 34078720);      //    524,288 B  (B+WR) bf16
  float*          av   = (float*)        (ws + 34603008);       //  2,097,152 B  scaled x@Aw^T

  cvt_bf16_kernel<<<(D_OUT * D_IN / 4) / 256, 256, 0, stream>>>(
      W_base, (unsigned int*)wbf, D_OUT * D_IN / 4);
  addcvt_bf16_kernel<<<(RANK * D_IN / 4) / 256, 256, 0, stream>>>(
      A, WL, (unsigned int*)awbf, RANK * D_IN / 4);
  addcvt_bf16_kernel<<<(D_OUT * RANK / 4) / 256, 256, 0, stream>>>(
      B, WR, (unsigned int*)bwbf, D_OUT * RANK / 4);
  lora_a_kernel<<<M_TOT / 128, 256, 0, stream>>>(x, awbf, av);
  main_gemm_kernel<<<dim3(D_OUT / BN, M_TOT / BM), 256, 0, stream>>>(
      x, wbf, b_base, av, bwbf, out);
}
